// MAGNOLayer_36258113912892
// MI455X (gfx1250) — matrix operand order repaired and checked by reading, not compiled, not run on hardware
//
#include <hip/hip_runtime.h>
#include <hip/hip_bf16.h>
#include <math.h>

typedef __attribute__((ext_vector_type(16))) _Float16 v16h;
typedef __attribute__((ext_vector_type(8)))  _Float16 v8h;
typedef __attribute__((ext_vector_type(4)))  _Float16 v4h;
typedef __attribute__((ext_vector_type(8)))  float    v8f;

#define HIDC 128
#define EDGES_PER_WAVE 32

// ---------------- helpers ----------------

static __device__ __forceinline__ unsigned enc_f32(float f) {
  unsigned u = __float_as_uint(f);
  return (u & 0x80000000u) ? ~u : (u | 0x80000000u);   // monotone float->uint
}
static __device__ __forceinline__ float dec_f32(unsigned c) {
  unsigned u = (c & 0x80000000u) ? (c & 0x7fffffffu) : ~c;
  return __uint_as_float(u);
}

static __device__ __forceinline__ void atomAddF(float* p, float v) {
  __hip_atomic_fetch_add(p, v, __ATOMIC_RELAXED, __HIP_MEMORY_SCOPE_AGENT);
}

static __device__ __forceinline__ float gelu_tanh(float x) {
  // jax.nn.gelu(approximate=True)
  float x3 = x * x * x;
  float t  = tanhf(0.7978845608028654f * fmaf(0.044715f, x3, x));
  return 0.5f * x * (1.0f + t);
}

static __device__ __forceinline__ v8f wmma16(v16h a, v16h b, v8f c) {
  return __builtin_amdgcn_wmma_f32_16x16x32_f16(false, a, false, b, (short)0, c,
                                                false, false);
}

// A fragment from row-major f16 matrix (ld=128): two contiguous 16B loads.
// lane l: row m=l&15, half h=l>>4; elems 0..7 -> K=k0+8h+j ; 8..15 -> K=k0+16+8h+j
static __device__ __forceinline__ v16h load_a16(const _Float16* A, int row0,
                                                int k0, int lane) {
  int m = lane & 15, h = lane >> 4;
  const _Float16* p = A + (size_t)(row0 + m) * HIDC + k0 + 8 * h;
  v8h lo = *(const v8h*)p;
  v8h hi = *(const v8h*)(p + 16);
  return __builtin_shufflevector(lo, hi, 0, 1, 2, 3, 4, 5, 6, 7, 8, 9, 10, 11,
                                 12, 13, 14, 15);
}

// B fragment from fragment-major f16 weight buffer: one contiguous 32B load.
static __device__ __forceinline__ v16h load_b16(const _Float16* Wfrag, int kk,
                                                int t, int lane) {
  return *(const v16h*)(Wfrag + ((size_t)(kk * 8 + t) * 32 + lane) * 16);
}

// ---------------- kernels ----------------

__global__ void magno_zero_f32(float* p, long n) {
  long i = (long)blockIdx.x * blockDim.x + threadIdx.x;
  if (i < n) p[i] = 0.0f;
}

__global__ void magno_cvt_f16(const float* __restrict__ src,
                              _Float16* __restrict__ dst, long n) {
  long i = (long)blockIdx.x * blockDim.x + threadIdx.x;
  if (i < n) dst[i] = (_Float16)src[i];
}

// Re-pack one 128x128 fp32 weight into f16 WMMA-B fragment order:
// dst[(((kt*8)+t)*32 + lane)*16 + j] = W[(kt*32 + 8*(lane>>4) + j + (j>=8 ? 8 : 0))*128 + t*16 + (lane&15)]
__global__ void magno_cvt_wfrag(const float* __restrict__ W,
                                _Float16* __restrict__ dst) {
  int tid = blockIdx.x * blockDim.x + threadIdx.x;
  if (tid >= 1024) return;
  int kt = tid >> 8, t = (tid >> 5) & 7, lane = tid & 31;
  int n = lane & 15, h = lane >> 4;
  int kbase = kt * 32 + 8 * h;
  _Float16* o = dst + ((size_t)(kt * 8 + t) * 32 + lane) * 16;
#pragma unroll
  for (int j = 0; j < 16; ++j)
    o[j] = (_Float16)W[(size_t)(kbase + j + (j >= 8 ? 8 : 0)) * HIDC + t * 16 + n];
}

__global__ void magno_prep_qnorm(const float* __restrict__ coords,
                                 float* __restrict__ qn, int n) {
  int i = blockIdx.x * blockDim.x + threadIdx.x;
  if (i >= n) return;
  float x = coords[2 * i + 0], y = coords[2 * i + 1];
  float nm = fmaxf(sqrtf(x * x + y * y), 1e-9f);
  qn[2 * i + 0] = x / nm;
  qn[2 * i + 1] = y / nm;
}

__global__ void magno_init_stats(unsigned* __restrict__ segmax,
                                 float* __restrict__ segsum, int n) {
  int i = blockIdx.x * blockDim.x + threadIdx.x;
  if (i >= n) return;
  segmax[i] = 0u;      // encoded -inf
  segsum[i] = 0.0f;
}

// out16 = A16 @ Wfrag   ([n_tiles*16,128] x [128,128], f16 in/out, f32 accum)
__global__ void magno_gemm16(const _Float16* __restrict__ A16,
                             const _Float16* __restrict__ Wfrag,
                             _Float16* __restrict__ out16, int n_tiles) {
  int lane = threadIdx.x & 31;
  int tile = blockIdx.x * (blockDim.x >> 5) + (threadIdx.x >> 5);
  if (tile >= n_tiles) return;            // wave-uniform branch
  int row0 = tile * 16;

  v8f acc[8];
#pragma unroll
  for (int t = 0; t < 8; ++t)
#pragma unroll
    for (int r = 0; r < 8; ++r) acc[t][r] = 0.0f;

#pragma unroll
  for (int kk = 0; kk < 4; ++kk) {
    v16h af = load_a16(A16, row0, kk * 32, lane);
#pragma unroll
    for (int t = 0; t < 8; ++t)
      acc[t] = wmma16(af, load_b16(Wfrag, kk, t, lane), acc[t]);
  }

  int n = lane & 15, h = lane >> 4;
#pragma unroll
  for (int t = 0; t < 8; ++t)
#pragma unroll
    for (int r = 0; r < 8; ++r)
      out16[(size_t)(row0 + r + 8 * h) * HIDC + t * 16 + n] =
          (_Float16)acc[t][r];
}

__global__ void magno_score(const float* __restrict__ qn,
                            const int* __restrict__ idx,
                            const int* __restrict__ seg,
                            float* __restrict__ scores,
                            unsigned* __restrict__ segmax, int E) {
  int e = blockIdx.x * blockDim.x + threadIdx.x;
  if (e >= E) return;
  int s = seg[e], i = idx[e];
  float sc = qn[2 * s] * qn[2 * i] + qn[2 * s + 1] * qn[2 * i + 1];
  scores[e] = sc;
  atomicMax(&segmax[s], enc_f32(sc));
}

__global__ void magno_expsum(float* __restrict__ scores,
                             const int* __restrict__ seg,
                             const unsigned* __restrict__ segmax,
                             float* __restrict__ segsum,
                             float* __restrict__ cnt, int E) {
  int e = blockIdx.x * blockDim.x + threadIdx.x;
  if (e >= E) return;
  int s = seg[e];
  float ev = expf(scores[e] - dec_f32(segmax[s]));
  scores[e] = ev;                 // store exp(score - max) in place
  atomAddF(&segsum[s], ev);
  atomAddF(&cnt[s], 1.0f);
}

// H[seg] += gelu(a * Y[idx] + b1). One wave walks EDGES_PER_WAVE sorted edges,
// register-accumulating while the segment id is unchanged (segids are sorted).
__global__ void magno_message(const _Float16* __restrict__ Y16,
                              const float* __restrict__ b1,
                              const float* __restrict__ escore,
                              const int* __restrict__ idx,
                              const int* __restrict__ seg,
                              const float* __restrict__ segsum,
                              float* __restrict__ H, int E) {
  int lane = threadIdx.x & 31;
  long wave = (long)blockIdx.x * (blockDim.x >> 5) + (threadIdx.x >> 5);
  long e0 = wave * EDGES_PER_WAVE;
  if (e0 >= E) return;
  long e1 = e0 + EDGES_PER_WAVE;
  if (e1 > E) e1 = E;

  float b1v[4];
#pragma unroll
  for (int c = 0; c < 4; ++c) b1v[c] = b1[lane * 4 + c];

  float acc[4] = {0.f, 0.f, 0.f, 0.f};
  int cur = -1;
  for (long e = e0; e < e1; ++e) {
    int s = seg[e];
    if (s != cur) {
      if (cur >= 0) {
#pragma unroll
        for (int c = 0; c < 4; ++c)
          atomAddF(&H[(size_t)cur * HIDC + lane * 4 + c], acc[c]);
      }
#pragma unroll
      for (int c = 0; c < 4; ++c) acc[c] = 0.0f;
      cur = s;
    }
    int i = idx[e];
    if (e + 1 < e1)
      __builtin_prefetch(Y16 + (size_t)idx[e + 1] * HIDC + lane * 4, 0, 0);
    float a = escore[e] / segsum[s];
    v4h y = *(const v4h*)(Y16 + (size_t)i * HIDC + lane * 4);
#pragma unroll
    for (int c = 0; c < 4; ++c)
      acc[c] += gelu_tanh(fmaf(a, (float)y[c], b1v[c]));
  }
  if (cur >= 0) {
#pragma unroll
    for (int c = 0; c < 4; ++c)
      atomAddF(&H[(size_t)cur * HIDC + lane * 4 + c], acc[c]);
  }
}

// out = LN( Ha@W2a + Hb@W2b + cnt1*b2a + cnt2*b2b ) * ln_scale + ln_bias
__global__ void magno_final(const _Float16* __restrict__ Ha16,
                            const _Float16* __restrict__ Hb16,
                            const _Float16* __restrict__ W2aF,
                            const _Float16* __restrict__ W2bF,
                            const float* __restrict__ b2a,
                            const float* __restrict__ b2b,
                            const float* __restrict__ cnt1,
                            const float* __restrict__ cnt2,
                            const float* __restrict__ lns,
                            const float* __restrict__ lnb,
                            float* __restrict__ out, int n_tiles) {
  int lane = threadIdx.x & 31;
  int tile = blockIdx.x * (blockDim.x >> 5) + (threadIdx.x >> 5);
  if (tile >= n_tiles) return;           // wave-uniform
  int row0 = tile * 16;

  v8f acc[8];
#pragma unroll
  for (int t = 0; t < 8; ++t)
#pragma unroll
    for (int r = 0; r < 8; ++r) acc[t][r] = 0.0f;

#pragma unroll
  for (int kk = 0; kk < 4; ++kk) {
    v16h afa = load_a16(Ha16, row0, kk * 32, lane);
    v16h afb = load_a16(Hb16, row0, kk * 32, lane);
#pragma unroll
    for (int t = 0; t < 8; ++t) {
      acc[t] = wmma16(afa, load_b16(W2aF, kk, t, lane), acc[t]);
      acc[t] = wmma16(afb, load_b16(W2bF, kk, t, lane), acc[t]);
    }
  }

  int n = lane & 15, h = lane >> 4;
  float c1[8], c2[8];
#pragma unroll
  for (int r = 0; r < 8; ++r) {
    int row = row0 + r + 8 * h;
    c1[r] = cnt1[row];
    c2[r] = cnt2[row];
  }
#pragma unroll
  for (int t = 0; t < 8; ++t) {
    float ba = b2a[t * 16 + n], bb = b2b[t * 16 + n];
#pragma unroll
    for (int r = 0; r < 8; ++r)
      acc[t][r] += fmaf(c1[r], ba, c2[r] * bb);
  }

  // layernorm across the 128 columns of each row: reduce over 8 reg tiles,
  // then across the 16 lanes of each half (xor masks < 16 stay in-half).
  float sum[8], sq[8];
#pragma unroll
  for (int r = 0; r < 8; ++r) { sum[r] = 0.f; sq[r] = 0.f; }
#pragma unroll
  for (int t = 0; t < 8; ++t)
#pragma unroll
    for (int r = 0; r < 8; ++r) {
      float v = acc[t][r];
      sum[r] += v;
      sq[r]  += v * v;
    }
#pragma unroll
  for (int mask = 1; mask < 16; mask <<= 1) {
#pragma unroll
    for (int r = 0; r < 8; ++r) {
      sum[r] += __shfl_xor(sum[r], mask, 32);
      sq[r]  += __shfl_xor(sq[r],  mask, 32);
    }
  }
  float mu[8], rstd[8];
#pragma unroll
  for (int r = 0; r < 8; ++r) {
    mu[r] = sum[r] * (1.0f / 128.0f);
    float var = sq[r] * (1.0f / 128.0f) - mu[r] * mu[r];
    rstd[r] = rsqrtf(var + 1e-6f);
  }
#pragma unroll
  for (int t = 0; t < 8; ++t) {
    int col = t * 16 + n;
    float g = lns[col], bt = lnb[col];
#pragma unroll
    for (int r = 0; r < 8; ++r) {
      size_t o = (size_t)(row0 + r + 8 * h) * HIDC + col;
      out[o] = fmaf((acc[t][r] - mu[r]) * rstd[r], g, bt);
    }
  }
}

// ---------------- launch ----------------

extern "C" void kernel_launch(void* const* d_in, const int* in_sizes, int n_in,
                              void* d_out, int out_size, void* d_ws,
                              size_t ws_size, hipStream_t stream) {
  const float* coords = (const float*)d_in[0];
  const float* feats  = (const float*)d_in[1];
  const float* W1a = (const float*)d_in[2];
  const float* b1a = (const float*)d_in[3];
  const float* W2a = (const float*)d_in[4];
  const float* b2a = (const float*)d_in[5];
  const float* W1b = (const float*)d_in[6];
  const float* b1b = (const float*)d_in[7];
  const float* W2b = (const float*)d_in[8];
  const float* b2b = (const float*)d_in[9];
  const float* lns = (const float*)d_in[10];
  const float* lnb = (const float*)d_in[11];
  const int* idx1 = (const int*)d_in[12];
  const int* seg1 = (const int*)d_in[13];
  const int* idx2 = (const int*)d_in[14];
  const int* seg2 = (const int*)d_in[15];

  const int N  = in_sizes[0] / 2;
  const int E1 = in_sizes[12];
  const int E2 = in_sizes[14];
  const int Emax = E1 > E2 ? E1 : E2;
  float* out = (float*)d_out;

  // workspace carve-out
  char* ws = (char*)d_ws;
  size_t off = 0;
  auto carve = [&](size_t bytes) -> void* {
    void* p = ws + off;
    off += (bytes + 255) & ~(size_t)255;
    return p;
  };
  float*     qn      = (float*)carve((size_t)N * 2 * 4);
  _Float16*  feats16 = (_Float16*)carve((size_t)N * HIDC * 2);
  _Float16*  Ya16    = (_Float16*)carve((size_t)N * HIDC * 2);
  _Float16*  Yb16    = (_Float16*)carve((size_t)N * HIDC * 2);
  float*     Ha      = (float*)carve((size_t)N * HIDC * 4);
  float*     Hb      = (float*)carve((size_t)N * HIDC * 4);
  _Float16*  Ha16    = (_Float16*)carve((size_t)N * HIDC * 2);
  _Float16*  Hb16    = (_Float16*)carve((size_t)N * HIDC * 2);
  _Float16*  W1aF    = (_Float16*)carve((size_t)HIDC * HIDC * 2);
  _Float16*  W1bF    = (_Float16*)carve((size_t)HIDC * HIDC * 2);
  _Float16*  W2aF    = (_Float16*)carve((size_t)HIDC * HIDC * 2);
  _Float16*  W2bF    = (_Float16*)carve((size_t)HIDC * HIDC * 2);
  unsigned*  segmax  = (unsigned*)carve((size_t)N * 4);
  float*     segsum  = (float*)carve((size_t)N * 4);
  float*     cnt1    = (float*)carve((size_t)N * 4);
  float*     cnt2    = (float*)carve((size_t)N * 4);
  float*     scores  = (float*)carve((size_t)Emax * 4);

  const int B = 256;
  const long NH = (long)N * HIDC;
  const int n_tiles = N / 16;           // N=50000 -> 3125 exact
  dim3 gemm_grid((n_tiles + 3) / 4);    // 4 waves / block
  dim3 gemm_block(128);

  // 0. zero accumulators / counts (fresh every call)
  magno_zero_f32<<<dim3((NH + B - 1) / B), dim3(B), 0, stream>>>(Ha, NH);
  magno_zero_f32<<<dim3((NH + B - 1) / B), dim3(B), 0, stream>>>(Hb, NH);
  magno_zero_f32<<<dim3((N + B - 1) / B), dim3(B), 0, stream>>>(cnt1, N);
  magno_zero_f32<<<dim3((N + B - 1) / B), dim3(B), 0, stream>>>(cnt2, N);

  // 1. normalized coords + f16 operand prep
  magno_prep_qnorm<<<dim3((N + B - 1) / B), dim3(B), 0, stream>>>(coords, qn, N);
  magno_cvt_f16<<<dim3((NH + B - 1) / B), dim3(B), 0, stream>>>(feats, feats16, NH);
  magno_cvt_wfrag<<<dim3(4), dim3(B), 0, stream>>>(W1a, W1aF);
  magno_cvt_wfrag<<<dim3(4), dim3(B), 0, stream>>>(W1b, W1bF);
  magno_cvt_wfrag<<<dim3(4), dim3(B), 0, stream>>>(W2a, W2aF);
  magno_cvt_wfrag<<<dim3(4), dim3(B), 0, stream>>>(W2b, W2bF);

  // 2. Y{a,b} = feats @ W1{a,b}  (WMMA, f16 out)
  magno_gemm16<<<gemm_grid, gemm_block, 0, stream>>>(feats16, W1aF, Ya16, n_tiles);
  magno_gemm16<<<gemm_grid, gemm_block, 0, stream>>>(feats16, W1bF, Yb16, n_tiles);

  // 3. edge set 1: softmax weights + message accumulation
  magno_init_stats<<<dim3((N + B - 1) / B), dim3(B), 0, stream>>>(segmax, segsum, N);
  magno_score<<<dim3((E1 + B - 1) / B), dim3(B), 0, stream>>>(qn, idx1, seg1,
                                                              scores, segmax, E1);
  magno_expsum<<<dim3((E1 + B - 1) / B), dim3(B), 0, stream>>>(scores, seg1, segmax,
                                                               segsum, cnt1, E1);
  {
    long waves = ((long)E1 + EDGES_PER_WAVE - 1) / EDGES_PER_WAVE;
    magno_message<<<dim3((unsigned)((waves + 7) / 8)), dim3(B), 0, stream>>>(
        Ya16, b1a, scores, idx1, seg1, segsum, Ha, E1);
  }

  // 4. edge set 2 (reuses score/stat buffers)
  magno_init_stats<<<dim3((N + B - 1) / B), dim3(B), 0, stream>>>(segmax, segsum, N);
  magno_score<<<dim3((E2 + B - 1) / B), dim3(B), 0, stream>>>(qn, idx2, seg2,
                                                              scores, segmax, E2);
  magno_expsum<<<dim3((E2 + B - 1) / B), dim3(B), 0, stream>>>(scores, seg2, segmax,
                                                               segsum, cnt2, E2);
  {
    long waves = ((long)E2 + EDGES_PER_WAVE - 1) / EDGES_PER_WAVE;
    magno_message<<<dim3((unsigned)((waves + 7) / 8)), dim3(B), 0, stream>>>(
        Yb16, b1b, scores, idx2, seg2, segsum, Hb, E2);
  }

  // 5. H -> f16, then fused dual GEMM + bias + layernorm (WMMA)
  magno_cvt_f16<<<dim3((NH + B - 1) / B), dim3(B), 0, stream>>>(Ha, Ha16, NH);
  magno_cvt_f16<<<dim3((NH + B - 1) / B), dim3(B), 0, stream>>>(Hb, Hb16, NH);
  magno_final<<<gemm_grid, gemm_block, 0, stream>>>(Ha16, Hb16, W2aF, W2bF,
                                                    b2a, b2b, cnt1, cnt2,
                                                    lns, lnb, out, n_tiles);
}
